// V_MambaBlock_82626580840705
// MI455X (gfx1250) — matrix operand order repaired and checked by reading, not compiled, not run on hardware
//
#include <hip/hip_runtime.h>

typedef __attribute__((ext_vector_type(16))) _Float16 v16h;
typedef __attribute__((ext_vector_type(8)))  float    v8f;

#define DIMC   96
#define DINNER 192
#define DSTATE 16
#define DTRANK 6
#define DCONV  4
#define BATCH  8
#define LSEQ   4096
#define NROWS  (BATCH * LSEQ)   /* 32768 */
#define XZC    384              /* 2*DINNER */
#define W2C    224              /* 192 dt cols + 16 B + 16 C */
#define NSEG   64               /* scan segments per batch */
#define TSEG   (LSEQ / NSEG)    /* 64 timesteps per segment */

// ---------------- WMMA fragment helpers (wave32, 16x16x32 f16) ----------------
// A-matrix 16x32 f16: lane&15 = M row; lane>=16 selects K-halves {8..15,24..31}.
// VGPR g<4 holds K = hs*8 + 2g + h ; g>=4 holds K = 16 + hs*8 + 2(g-4) + h.
__device__ __forceinline__ v16h load_a_frag_lds(const _Float16* a, int lda, int kbase, int lane) {
  v16h f;
  const int m  = lane & 15;
  const int hs = (lane >> 4) & 1;
#pragma unroll
  for (int e = 0; e < 16; ++e) {
    const int g  = e >> 1;
    const int hh = e & 1;
    const int k  = (g < 4) ? (hs * 8 + g * 2 + hh) : (16 + hs * 8 + (g - 4) * 2 + hh);
    f[e] = a[m * lda + kbase + k];
  }
  return f;
}

// B-matrix 32x16 f16 (row-major K x N in memory): lane&15 = N column; K order matches A fragment.
__device__ __forceinline__ v16h load_b_frag(const _Float16* bw, int ldb, int kbase, int n0, int lane) {
  v16h f;
  const int n  = n0 + (lane & 15);
  const int hs = (lane >> 4) & 1;
#pragma unroll
  for (int e = 0; e < 16; ++e) {
    const int k = (e < 8) ? (hs * 8 + e) : (16 + hs * 8 + (e - 8));
    f[e] = bw[(size_t)(kbase + k) * ldb + n];
  }
  return f;
}

__device__ __forceinline__ float softplus_f(float v) {
  return (v > 20.f) ? v : log1pf(__expf(v));
}
__device__ __forceinline__ float silu_f(float v) {
  return v / (1.f + __expf(-v));
}

// ---------------- Kernel 1: weight prep ----------------
__global__ __launch_bounds__(256) void mamba_prep_kernel(
    const float* __restrict__ w_in, const float* __restrict__ w_xproj,
    const float* __restrict__ w_dt, const float* __restrict__ w_out,
    const float* __restrict__ a_log,
    _Float16* __restrict__ w_in_h, _Float16* __restrict__ W2h,
    _Float16* __restrict__ w_out_h, float* __restrict__ Aneg) {
  const int n1 = DIMC * XZC;
  const int n2 = DINNER * DINNER;
  const int n3 = DINNER * 32;
  const int n4 = DINNER * DIMC;
  const int n5 = DINNER * DSTATE;
  const int total = n1 + n2 + n3 + n4 + n5;
  for (int i = blockIdx.x * blockDim.x + threadIdx.x; i < total; i += gridDim.x * blockDim.x) {
    if (i < n1) {
      w_in_h[i] = (_Float16)w_in[i];
    } else if (i < n1 + n2) {
      const int j = i - n1;
      const int k = j / DINNER, col = j % DINNER;
      float s = 0.f;
#pragma unroll
      for (int r = 0; r < DTRANK; ++r)
        s += w_xproj[k * (DTRANK + 2 * DSTATE) + r] * w_dt[r * DINNER + col];
      W2h[k * W2C + col] = (_Float16)s;
    } else if (i < n1 + n2 + n3) {
      const int j = i - n1 - n2;
      const int k = j / 32, n = j % 32;
      W2h[k * W2C + DINNER + n] = (_Float16)w_xproj[k * (DTRANK + 2 * DSTATE) + DTRANK + n];
    } else if (i < n1 + n2 + n3 + n4) {
      const int j = i - n1 - n2 - n3;
      w_out_h[j] = (_Float16)w_out[j];
    } else {
      const int j = i - n1 - n2 - n3 - n4;
      Aneg[j] = -__expf(a_log[j]);
    }
  }
}

// ---------------- Kernel 2: LayerNorm + in-projection GEMM (WMMA) ----------------
__global__ __launch_bounds__(128) void mamba_ln_inproj_kernel(
    const float* __restrict__ x, const float* __restrict__ gam, const float* __restrict__ bet,
    const _Float16* __restrict__ w_in_h, _Float16* __restrict__ xz) {
  __shared__ _Float16 aT[16 * DIMC];
  const int row0 = blockIdx.x * 16;
  const int tid = threadIdx.x;
  if (tid < 16) {
    const int row = row0 + tid;
    const int b = row / LSEQ, l = row % LSEQ;
    const float* xp = x + (size_t)b * DIMC * LSEQ + l;
    float mu = 0.f;
#pragma unroll 8
    for (int c = 0; c < DIMC; ++c) mu += xp[(size_t)c * LSEQ];
    mu *= (1.f / DIMC);
    float var = 0.f;
#pragma unroll 8
    for (int c = 0; c < DIMC; ++c) { float d = xp[(size_t)c * LSEQ] - mu; var += d * d; }
    var *= (1.f / DIMC);
    const float rs = rsqrtf(var + 1e-5f);
#pragma unroll 8
    for (int c = 0; c < DIMC; ++c)
      aT[tid * DIMC + c] = (_Float16)((xp[(size_t)c * LSEQ] - mu) * rs * gam[c] + bet[c]);
  }
  __syncthreads();
  const int wave = tid >> 5, lane = tid & 31;
  for (int nt = wave; nt < XZC / 16; nt += 4) {
    v8f acc = {};
#pragma unroll
    for (int kt = 0; kt < DIMC / 32; ++kt) {
      v16h af = load_a_frag_lds(aT, DIMC, kt * 32, lane);
      v16h bf = load_b_frag(w_in_h, XZC, kt * 32, nt * 16, lane);
      acc = __builtin_amdgcn_wmma_f32_16x16x32_f16(false, af, false, bf, (short)0, acc, false, false);
    }
    const int n = nt * 16 + (lane & 15);
    const int mb = (lane >> 4) * 8;
#pragma unroll
    for (int r = 0; r < 8; ++r)
      xz[(size_t)(row0 + mb + r) * XZC + n] = (_Float16)acc[r];
  }
}

// ---------------- Kernel 3: causal depthwise conv (k=4) + SiLU ----------------
__global__ __launch_bounds__(256) void mamba_conv_silu_kernel(
    const _Float16* __restrict__ xz, const float* __restrict__ cw,
    const float* __restrict__ cb, float* __restrict__ xconv) {
  const int total = NROWS * DINNER;
  for (int i = blockIdx.x * blockDim.x + threadIdx.x; i < total; i += gridDim.x * blockDim.x) {
    const int d  = i % DINNER;
    const int bl = i / DINNER;
    const int l  = bl % LSEQ;
    float s = cb[d];
#pragma unroll
    for (int k = 0; k < DCONV; ++k) {
      const int lk = l - (DCONV - 1) + k;
      if (lk >= 0)
        s += cw[d * DCONV + k] * (float)xz[(size_t)(bl - l + lk) * XZC + d];
    }
    xconv[i] = silu_f(s);
  }
}

// ---------------- Kernel 4: x-projection GEMM (WMMA) + softplus epilogue ----------------
__global__ __launch_bounds__(128) void mamba_xproj_kernel(
    const float* __restrict__ xconv, const _Float16* __restrict__ W2h,
    const float* __restrict__ b_dt, float* __restrict__ dt,
    float* __restrict__ Bm, float* __restrict__ Cm) {
  __shared__ _Float16 aT[16 * DINNER];
  const int row0 = blockIdx.x * 16;
  for (int i = threadIdx.x; i < 16 * DINNER; i += 128) {
    const int m = i / DINNER, k = i % DINNER;
    aT[i] = (_Float16)xconv[(size_t)(row0 + m) * DINNER + k];
  }
  __syncthreads();
  const int wave = threadIdx.x >> 5, lane = threadIdx.x & 31;
  for (int nt = wave; nt < W2C / 16; nt += 4) {
    v8f acc = {};
#pragma unroll
    for (int kt = 0; kt < DINNER / 32; ++kt) {
      v16h af = load_a_frag_lds(aT, DINNER, kt * 32, lane);
      v16h bf = load_b_frag(W2h, W2C, kt * 32, nt * 16, lane);
      acc = __builtin_amdgcn_wmma_f32_16x16x32_f16(false, af, false, bf, (short)0, acc, false, false);
    }
    const int n = nt * 16 + (lane & 15);
    const int mb = (lane >> 4) * 8;
#pragma unroll
    for (int r = 0; r < 8; ++r) {
      const size_t row = (size_t)(row0 + mb + r);
      const float v = acc[r];
      if (n < DINNER)            dt[row * DINNER + n] = softplus_f(v + b_dt[n]);
      else if (n < DINNER + 16)  Bm[row * DSTATE + (n - DINNER)] = v;
      else                       Cm[row * DSTATE + (n - DINNER - 16)] = v;
    }
  }
}

// ---------------- Kernel 5a: chunked scan, pass 1 (segment-local) ----------------
// Block = (b, segment); thread = channel d. Computes local scan from h=0 and Σdt.
__global__ __launch_bounds__(DINNER) void mamba_scan_pass1(
    const float* __restrict__ dt, const float* __restrict__ Bm,
    const float* __restrict__ xconv, const float* __restrict__ Aneg,
    float* __restrict__ sumdt, float* __restrict__ hloc) {
  __shared__ float sB[TSEG * DSTATE];
  const int b = blockIdx.x / NSEG;
  const int s = blockIdx.x % NSEG;
  const int d = threadIdx.x;
  const int t0 = s * TSEG;
  for (int i = threadIdx.x; i < TSEG * DSTATE; i += DINNER)
    sB[i] = Bm[(size_t)(b * LSEQ + t0 + i / DSTATE) * DSTATE + (i % DSTATE)];
  __syncthreads();
  float a[DSTATE], h[DSTATE];
#pragma unroll
  for (int n = 0; n < DSTATE; ++n) { a[n] = Aneg[d * DSTATE + n]; h[n] = 0.f; }
  float sd = 0.f;
  for (int tt = 0; tt < TSEG; ++tt) {
    const size_t row = (size_t)b * LSEQ + t0 + tt;
    const float dtv = dt[row * DINNER + d];
    const float xv  = xconv[row * DINNER + d];
    const float dx  = dtv * xv;
    sd += dtv;
#pragma unroll
    for (int n = 0; n < DSTATE; ++n)
      h[n] = __expf(dtv * a[n]) * h[n] + dx * sB[tt * DSTATE + n];
  }
  const size_t base = (size_t)(b * NSEG + s) * DINNER + d;
  sumdt[base] = sd;
#pragma unroll
  for (int n = 0; n < DSTATE; ++n) hloc[base * DSTATE + n] = h[n];
}

// ---------------- Kernel 5b: prefix fix-up across segments ----------------
// One thread per (b,d,n): sequential prefix over only NSEG=64 segments.
__global__ __launch_bounds__(256) void mamba_scan_fix(
    const float* __restrict__ sumdt, const float* __restrict__ hloc,
    const float* __restrict__ Aneg, float* __restrict__ hstart) {
  const int idx = blockIdx.x * blockDim.x + threadIdx.x;
  if (idx >= BATCH * DINNER * DSTATE) return;
  const int n = idx % DSTATE;
  const int d = (idx / DSTATE) % DINNER;
  const int b = idx / (DSTATE * DINNER);
  const float a = Aneg[d * DSTATE + n];
  float h = 0.f;
  for (int s = 0; s < NSEG; ++s) {
    const size_t base = (size_t)(b * NSEG + s) * DINNER + d;
    hstart[base * DSTATE + n] = h;
    h = __expf(a * sumdt[base]) * h + hloc[base * DSTATE + n];
  }
}

// ---------------- Kernel 5c: chunked scan, pass 2 (+ D-skip + SiLU(z) gate) ----------
__global__ __launch_bounds__(DINNER) void mamba_scan_pass2(
    const float* __restrict__ dt, const float* __restrict__ Bm, const float* __restrict__ Cm,
    const float* __restrict__ xconv, const _Float16* __restrict__ xz,
    const float* __restrict__ Aneg, const float* __restrict__ d_skip,
    const float* __restrict__ hstart, float* __restrict__ yg) {
  __shared__ float sB[TSEG * DSTATE];
  __shared__ float sC[TSEG * DSTATE];
  const int b = blockIdx.x / NSEG;
  const int s = blockIdx.x % NSEG;
  const int d = threadIdx.x;
  const int t0 = s * TSEG;
  for (int i = threadIdx.x; i < TSEG * DSTATE; i += DINNER) {
    const size_t r = (size_t)(b * LSEQ + t0 + i / DSTATE);
    sB[i] = Bm[r * DSTATE + (i % DSTATE)];
    sC[i] = Cm[r * DSTATE + (i % DSTATE)];
  }
  __syncthreads();
  const size_t base = (size_t)(b * NSEG + s) * DINNER + d;
  float a[DSTATE], h[DSTATE];
#pragma unroll
  for (int n = 0; n < DSTATE; ++n) {
    a[n] = Aneg[d * DSTATE + n];
    h[n] = hstart[base * DSTATE + n];
  }
  const float dsk = d_skip[d];
  for (int tt = 0; tt < TSEG; ++tt) {
    const size_t row = (size_t)b * LSEQ + t0 + tt;
    const float dtv = dt[row * DINNER + d];
    const float xv  = xconv[row * DINNER + d];
    const float dx  = dtv * xv;
    float y = 0.f;
#pragma unroll
    for (int n = 0; n < DSTATE; ++n) {
      h[n] = __expf(dtv * a[n]) * h[n] + dx * sB[tt * DSTATE + n];
      y += h[n] * sC[tt * DSTATE + n];
    }
    y += dsk * xv;
    const float z = (float)xz[row * XZC + DINNER + d];
    yg[row * DINNER + d] = y * silu_f(z);
  }
}

// ---------------- Kernel 6: out-projection GEMM (WMMA) + residual, NCHW store ----------
__global__ __launch_bounds__(128) void mamba_outproj_kernel(
    const float* __restrict__ yg, const _Float16* __restrict__ w_out_h,
    const float* __restrict__ x, float* __restrict__ out) {
  __shared__ _Float16 aT[16 * DINNER];
  const int row0 = blockIdx.x * 16;
  for (int i = threadIdx.x; i < 16 * DINNER; i += 128) {
    const int m = i / DINNER, k = i % DINNER;
    aT[i] = (_Float16)yg[(size_t)(row0 + m) * DINNER + k];
  }
  __syncthreads();
  const int wave = threadIdx.x >> 5, lane = threadIdx.x & 31;
  for (int nt = wave; nt < DIMC / 16; nt += 4) {
    v8f acc = {};
#pragma unroll
    for (int kt = 0; kt < DINNER / 32; ++kt) {
      v16h af = load_a_frag_lds(aT, DINNER, kt * 32, lane);
      v16h bf = load_b_frag(w_out_h, DIMC, kt * 32, nt * 16, lane);
      acc = __builtin_amdgcn_wmma_f32_16x16x32_f16(false, af, false, bf, (short)0, acc, false, false);
    }
    const int c = nt * 16 + (lane & 15);
    const int mb = (lane >> 4) * 8;
#pragma unroll
    for (int r = 0; r < 8; ++r) {
      const int row = row0 + mb + r;
      const int b = row / LSEQ, l = row % LSEQ;
      const size_t idx = (size_t)b * DIMC * LSEQ + (size_t)c * LSEQ + l;
      out[idx] = x[idx] + acc[r];
    }
  }
}

// ---------------- Launch ----------------
extern "C" void kernel_launch(void* const* d_in, const int* in_sizes, int n_in,
                              void* d_out, int out_size, void* d_ws, size_t ws_size,
                              hipStream_t stream) {
  const float* x        = (const float*)d_in[0];
  const float* ln_gamma = (const float*)d_in[1];
  const float* ln_beta  = (const float*)d_in[2];
  const float* w_in     = (const float*)d_in[3];
  const float* conv_w   = (const float*)d_in[4];
  const float* conv_b   = (const float*)d_in[5];
  const float* w_xproj  = (const float*)d_in[6];
  const float* w_dt     = (const float*)d_in[7];
  const float* b_dt     = (const float*)d_in[8];
  const float* a_log    = (const float*)d_in[9];
  const float* d_skip   = (const float*)d_in[10];
  const float* w_out    = (const float*)d_in[11];
  float* out = (float*)d_out;

  // Workspace layout (all offsets multiples of 256 bytes).
  char* ws = (char*)d_ws;
  size_t off = 0;
  _Float16* w_in_h  = (_Float16*)(ws + off); off += (size_t)DIMC * XZC * 2;
  _Float16* W2h     = (_Float16*)(ws + off); off += (size_t)DINNER * W2C * 2;
  _Float16* w_out_h = (_Float16*)(ws + off); off += (size_t)DINNER * DIMC * 2;
  float*    Aneg    = (float*)(ws + off);    off += (size_t)DINNER * DSTATE * 4;
  _Float16* xz      = (_Float16*)(ws + off); off += (size_t)NROWS * XZC * 2;
  float*    xconv   = (float*)(ws + off);    off += (size_t)NROWS * DINNER * 4;
  float*    dt      = (float*)(ws + off);    off += (size_t)NROWS * DINNER * 4;
  float*    Bmat    = (float*)(ws + off);    off += (size_t)NROWS * DSTATE * 4;
  float*    Cmat    = (float*)(ws + off);    off += (size_t)NROWS * DSTATE * 4;
  float*    yg      = (float*)(ws + off);    off += (size_t)NROWS * DINNER * 4;
  float*    sumdt   = (float*)(ws + off);    off += (size_t)BATCH * NSEG * DINNER * 4;
  float*    hloc    = (float*)(ws + off);    off += (size_t)BATCH * NSEG * DINNER * DSTATE * 4;
  float*    hstart  = (float*)(ws + off);    off += (size_t)BATCH * NSEG * DINNER * DSTATE * 4;
  (void)ws_size; (void)in_sizes; (void)n_in; (void)out_size;

  mamba_prep_kernel<<<396, 256, 0, stream>>>(w_in, w_xproj, w_dt, w_out, a_log,
                                             w_in_h, W2h, w_out_h, Aneg);
  mamba_ln_inproj_kernel<<<NROWS / 16, 128, 0, stream>>>(x, ln_gamma, ln_beta, w_in_h, xz);
  mamba_conv_silu_kernel<<<(NROWS * DINNER) / 256, 256, 0, stream>>>(xz, conv_w, conv_b, xconv);
  mamba_xproj_kernel<<<NROWS / 16, 128, 0, stream>>>(xconv, W2h, b_dt, dt, Bmat, Cmat);
  mamba_scan_pass1<<<BATCH * NSEG, DINNER, 0, stream>>>(dt, Bmat, xconv, Aneg, sumdt, hloc);
  mamba_scan_fix<<<(BATCH * DINNER * DSTATE + 255) / 256, 256, 0, stream>>>(sumdt, hloc, Aneg, hstart);
  mamba_scan_pass2<<<BATCH * NSEG, DINNER, 0, stream>>>(dt, Bmat, Cmat, xconv, xz, Aneg, d_skip,
                                                        hstart, yg);
  mamba_outproj_kernel<<<NROWS / 16, 128, 0, stream>>>(yg, w_out_h, x, out);
}
